// RNN_37580963840363
// MI455X (gfx1250) — compile-verified, matrix operand order checked by reading, not run on hardware
//
#include <hip/hip_runtime.h>
#include <hip/hip_bf16.h>
#include <math.h>

#define B_ 64
#define T_ 2048
#define I_ 256
#define H_ 512
#define O_ 256

typedef __attribute__((ext_vector_type(16))) __bf16        v16bf;
typedef __attribute__((ext_vector_type(8)))  float         v8f;
typedef __attribute__((ext_vector_type(4)))  unsigned int  v4u;
typedef int v4i_vs __attribute__((vector_size(16)));       // GCC-style, matches builtin proto
typedef __attribute__((address_space(1))) v4i_vs* gas_v4i;
typedef __attribute__((address_space(3))) v4i_vs* las_v4i;

#if __has_builtin(__builtin_amdgcn_tanhf)
#define TANHF(x) __builtin_amdgcn_tanhf(x)
#else
#define TANHF(x) tanhf(x)
#endif

__device__ inline v8f wmma_bf16(v16bf a, v16bf b, v8f c) {
  return __builtin_amdgcn_wmma_f32_16x16x32_bf16(false, a, false, b, (short)0, c,
                                                 false, false);
}

// ---- CDNA5 async global->LDS copy (ASYNCcnt-tracked), 16B per lane --------
__device__ inline void async_copy_b128(const float* gsrc, float* lds_dst) {
#if __has_builtin(__builtin_amdgcn_global_load_async_to_lds_b128)
  __builtin_amdgcn_global_load_async_to_lds_b128((gas_v4i)gsrc, (las_v4i)lds_dst,
                                                 0, 0);
#else
  unsigned lds_off =
      (unsigned)(unsigned long long)(__attribute__((address_space(3))) void*)lds_dst;
  asm volatile("global_load_async_to_lds_b128 %0, %1, off" ::"v"(lds_off),
               "v"((unsigned long long)(size_t)gsrc)
               : "memory");
#endif
}

template <int N> __device__ inline void wait_asynccnt() {
#if __has_builtin(__builtin_amdgcn_s_wait_asynccnt)
  __builtin_amdgcn_s_wait_asynccnt(N);
#else
  asm volatile("s_wait_asynccnt %0" ::"i"(N) : "memory");
#endif
}

// 16(row) x 32(K) bf16 fragment, element (r,k) at base[r*ld + k].
// Per ISA 7.12.2: lanes 0-15 hold K-half 0, lanes 16-31 hold K-half +8,
// bf16 pairs per dword. Works for A (r=M) and B (r=N) since weights are
// stored W[n][k] row-major (B^T -> contiguous K pairs).
__device__ inline v16bf load_frag_bf16(const __bf16* base, int ld) {
  const int lane = threadIdx.x & 31;
  const int r    = lane & 15;
  const int kh   = (lane >> 4) << 3;
  const __bf16* p = base + (size_t)r * ld + kh;
  v16bf f;
#pragma unroll
  for (int j = 0; j < 8; ++j) {
    const int k = ((j & 3) << 1) + ((j >> 2) << 4); // 0,2,4,6,16,18,20,22
    f[2 * j]     = p[k];
    f[2 * j + 1] = p[k + 1];
  }
  return f;
}

__device__ inline v16bf load_frag_f32(const float* base, long ld) {
  const int lane = threadIdx.x & 31;
  const int r    = lane & 15;
  const int kh   = (lane >> 4) << 3;
  const float* p = base + (long)r * ld + kh;
  v16bf f;
#pragma unroll
  for (int j = 0; j < 8; ++j) {
    const int k = ((j & 3) << 1) + ((j >> 2) << 4);
    f[2 * j]     = (__bf16)p[k];
    f[2 * j + 1] = (__bf16)p[k + 1];
  }
  return f;
}

__global__ void cvt_f32_to_bf16(const float* __restrict__ s,
                                __bf16* __restrict__ d, int n) {
  int i = blockIdx.x * blockDim.x + threadIdx.x;
  if (i < n) d[i] = (__bf16)s[i];
}

// ---------------------------------------------------------------------------
// Phase 1: xp[t,b,n] = sum_k x[b,t,k]*W_ih[n,k] + (b_ih[n]+b_hh[n])
// Grid: x = N-slices (fast, shares the A panel -> L2/L0 hits), y = timestep.
// ---------------------------------------------------------------------------
__global__ __launch_bounds__(256) void xproj_kernel(
    const float* __restrict__ x, const __bf16* __restrict__ Wih,
    const float* __restrict__ b_ih, const float* __restrict__ b_hh,
    float* __restrict__ xp) {
  const int t    = blockIdx.y;
  const int wave = threadIdx.x >> 5;
  const int lane = threadIdx.x & 31;
  const int b0   = (wave >> 1) << 4;                  // 0,16,32,48
  const int n0   = blockIdx.x * 32 + ((wave & 1) << 4);
  v8f c = {0.f, 0.f, 0.f, 0.f, 0.f, 0.f, 0.f, 0.f};
  const float* abase = x + ((size_t)b0 * T_ + t) * I_;
#pragma unroll
  for (int kt = 0; kt < I_; kt += 32) {
    v16bf a = load_frag_f32(abase + kt, (long)T_ * I_);
    v16bf b = load_frag_bf16(Wih + (size_t)n0 * I_ + kt, I_);
    c = wmma_bf16(a, b, c);
  }
  const int   n    = n0 + (lane & 15);
  const int   mb   = (lane >> 4) << 3;
  const float bias = b_ih[n] + b_hh[n];
#pragma unroll
  for (int v = 0; v < 8; ++v) {
    const int b = b0 + mb + v;
    xp[((size_t)t * B_ + b) * H_ + n] = c[v] + bias;
  }
}

// ---------------------------------------------------------------------------
// Phase 2: recurrence. 4 blocks x 16 batch rows (sync-free batch partition).
// 256 threads = 8 waves, each wave owns 64 N cols = 4 independent WMMA
// accumulator chains (covers the bf16 WMMA->WMMA hazard window).
// h slice (16x512 bf16) lives in LDS; xp[t] panels (32KB) are double-buffered
// in LDS via GLOBAL_LOAD_ASYNC_TO_LDS_B128, prefetched during the k-loop and
// fenced with s_wait_asynccnt. W_hh (512KB bf16) streams from L2 (hot).
// h_t is written back with a cooperative b128 LDS->global copy.
// ---------------------------------------------------------------------------
__global__ __launch_bounds__(256) void recur_kernel(
    const float* __restrict__ xp, const __bf16* __restrict__ Whh,
    __bf16* __restrict__ hist) {
  __shared__ __bf16 hbuf[16 * H_];          // 16 KB: h_{t-1} rows b0..b0+15
  __shared__ float  xpbuf[2][16 * H_];      // 2 x 32 KB: xp[t] double buffer
  const int b0   = blockIdx.x << 4;
  const int wave = threadIdx.x >> 5;        // 0..7
  const int lane = threadIdx.x & 31;
  const int n0   = wave << 6;               // 64 cols per wave

  for (int i = threadIdx.x; i < 16 * H_; i += blockDim.x) hbuf[i] = (__bf16)0.f;
  // Prefetch xp[0] into buffer 0 (8 x b128 per thread = 32 KB block panel).
  {
    const float* src = xp + (size_t)b0 * H_;
#pragma unroll
    for (int r = 0; r < 8; ++r) {
      const int c = r * 256 + threadIdx.x;
      async_copy_b128(src + c * 4, &xpbuf[0][c * 4]);
    }
  }
  __syncthreads();

  for (int t = 0; t < T_; ++t) {
    // Kick off async prefetch of xp[t+1] while we do the matmul for step t.
    if (t + 1 < T_) {
      const float* src = xp + ((size_t)(t + 1) * B_ + b0) * H_;
      float*       dst = xpbuf[(t + 1) & 1];
#pragma unroll
      for (int r = 0; r < 8; ++r) {
        const int c = r * 256 + threadIdx.x;
        async_copy_b128(src + c * 4, dst + c * 4);
      }
    }

    const v8f zero = {0.f, 0.f, 0.f, 0.f, 0.f, 0.f, 0.f, 0.f};
    v8f acc[4];
#pragma unroll
    for (int s = 0; s < 4; ++s) acc[s] = zero;
#pragma unroll 2
    for (int kt = 0; kt < H_; kt += 32) {
      v16bf a = load_frag_bf16(&hbuf[kt], H_);          // ds reads of h_{t-1}
#pragma unroll
      for (int s = 0; s < 4; ++s) {
        v16bf w = load_frag_bf16(Whh + (size_t)(n0 + s * 16) * H_ + kt, H_);
        acc[s] = wmma_bf16(a, w, acc[s]);
      }
    }

    // Async loads complete in-order per wave: after issuing 8 for t+1,
    // asynccnt<=8 guarantees the 8 for t (issued last iteration) are done.
    if (t + 1 < T_) wait_asynccnt<8>(); else wait_asynccnt<0>();
    __syncthreads();   // hbuf reads done everywhere + xp[t] panel visible

    const float* xpc = xpbuf[t & 1];
    const int    n   = lane & 15;
    const int    mb  = (lane >> 4) << 3;
#pragma unroll
    for (int s = 0; s < 4; ++s) {
      const int nn = n0 + s * 16 + n;
#pragma unroll
      for (int v = 0; v < 8; ++v) {
        const int m = mb + v;
        hbuf[m * H_ + nn] = (__bf16)TANHF(acc[s][v] + xpc[m * H_ + nn]);
      }
    }
    __syncthreads();   // h_t complete in LDS

    // Cooperative vectorized h_t write-back: 16 rows x 1KB, b128 chunks.
#pragma unroll
    for (int r = 0; r < 4; ++r) {
      const int chunk = r * 256 + threadIdx.x;   // 0..1023
      const int row   = chunk >> 6;
      const int off   = (chunk & 63) * 8;        // bf16 elements (16B)
      v4u d = *(const v4u*)&hbuf[row * H_ + off];
      *(v4u*)&hist[((size_t)(b0 + row) * T_ + t) * H_ + off] = d;
    }
  }
}

// ---------------------------------------------------------------------------
// Phase 3: out[r,n] = sum_k hist[r,k]*W_fc[n,k] + b_fc[n]
// Grid: x = N-slices (fast), y = row panels (A reuse across consecutive blocks)
// ---------------------------------------------------------------------------
__global__ __launch_bounds__(256) void fc_kernel(
    const __bf16* __restrict__ hist, const __bf16* __restrict__ Wfc,
    const float* __restrict__ b_fc, float* __restrict__ out) {
  const int    wave = threadIdx.x >> 5;
  const int    lane = threadIdx.x & 31;
  const size_t r0   = (size_t)blockIdx.y * 64 + ((wave >> 1) << 4);
  const int    n0   = blockIdx.x * 32 + ((wave & 1) << 4);
  v8f c = {0.f, 0.f, 0.f, 0.f, 0.f, 0.f, 0.f, 0.f};
#pragma unroll 4
  for (int kt = 0; kt < H_; kt += 32) {
    v16bf a = load_frag_bf16(hist + r0 * H_ + kt, H_);
    v16bf b = load_frag_bf16(Wfc + (size_t)n0 * H_ + kt, H_);
    c = wmma_bf16(a, b, c);
  }
  const int   n    = n0 + (lane & 15);
  const int   mb   = (lane >> 4) << 3;
  const float bias = b_fc[n];
#pragma unroll
  for (int v = 0; v < 8; ++v)
    out[(r0 + mb + v) * O_ + n] = c[v] + bias;
}

__global__ void hidden_kernel(const __bf16* __restrict__ hist,
                              float* __restrict__ hid) {
  int i = blockIdx.x * blockDim.x + threadIdx.x;
  if (i < B_ * H_) {
    int b = i / H_, h = i % H_;
    hid[i] = (float)hist[((size_t)b * T_ + (T_ - 1)) * H_ + h];
  }
}

extern "C" void kernel_launch(void* const* d_in, const int* in_sizes, int n_in,
                              void* d_out, int out_size, void* d_ws, size_t ws_size,
                              hipStream_t stream) {
  const float* x   = (const float*)d_in[0];
  const float* Wih = (const float*)d_in[1];
  const float* Whh = (const float*)d_in[2];
  const float* bih = (const float*)d_in[3];
  const float* bhh = (const float*)d_in[4];
  const float* Wfc = (const float*)d_in[5];
  const float* bfc = (const float*)d_in[6];

  char*  ws  = (char*)d_ws;
  size_t off = 0;
  float*  xp    = (float*)(ws + off);  off += (size_t)T_ * B_ * H_ * sizeof(float);
  __bf16* hist  = (__bf16*)(ws + off); off += (size_t)B_ * T_ * H_ * sizeof(__bf16);
  __bf16* WhhBf = (__bf16*)(ws + off); off += (size_t)H_ * H_ * sizeof(__bf16);
  __bf16* WihBf = (__bf16*)(ws + off); off += (size_t)H_ * I_ * sizeof(__bf16);
  __bf16* WfcBf = (__bf16*)(ws + off); off += (size_t)O_ * H_ * sizeof(__bf16);

  cvt_f32_to_bf16<<<(H_ * I_ + 255) / 256, 256, 0, stream>>>(Wih, WihBf, H_ * I_);
  cvt_f32_to_bf16<<<(H_ * H_ + 255) / 256, 256, 0, stream>>>(Whh, WhhBf, H_ * H_);
  cvt_f32_to_bf16<<<(O_ * H_ + 255) / 256, 256, 0, stream>>>(Wfc, WfcBf, O_ * H_);

  dim3 gx(H_ / 32, T_);
  xproj_kernel<<<gx, 256, 0, stream>>>(x, WihBf, bih, bhh, xp);

  recur_kernel<<<B_ / 16, 256, 0, stream>>>(xp, WhhBf, hist);

  dim3 gf(O_ / 32, (B_ * T_) / 64);
  fc_kernel<<<gf, 256, 0, stream>>>(hist, WfcBf, bfc, (float*)d_out);

  hidden_kernel<<<(B_ * H_ + 255) / 256, 256, 0, stream>>>(
      hist, (float*)d_out + (size_t)B_ * T_ * O_);
}
